// HoloLayer_64561948394136
// MI455X (gfx1250) — matrix operand order repaired and checked by reading, not compiled, not run on hardware
//
#include <hip/hip_runtime.h>
#include <math.h>

// ---------------------------------------------------------------------------
// Types for CDNA5 WMMA (wave32): v_wmma_f32_16x16x32_bf16
// ---------------------------------------------------------------------------
typedef __attribute__((ext_vector_type(16))) __bf16 v16bf;
typedef __attribute__((ext_vector_type(8)))  __bf16 v8bf;
typedef __attribute__((ext_vector_type(8)))  float  v8f;
typedef unsigned int v4u __attribute__((ext_vector_type(4)));
typedef int          v8i __attribute__((ext_vector_type(8)));
typedef int          v4i __attribute__((ext_vector_type(4)));

#define TM 128
#define TN 128
#define TK 32
#define LDS_STRIDE (TK + 8)   // 40 elems = 80B LDS rows (64B data + 16B pad)

enum { EPI_F32 = 0, EPI_ADD_F32 = 1, EPI_GELU_BF16 = 2 };

// Tensor Data Mover availability (probe-verified builtin names; arity differs
// between ROCm 7.2 clang-22 (5 args) and clang-23 toolchain (6 args)).
#if defined(__has_builtin)
#  if __has_builtin(__builtin_amdgcn_tensor_load_to_lds) && \
      __has_builtin(__builtin_amdgcn_s_wait_tensorcnt)
#    define USE_TDM 1
#  endif
#endif
#ifndef USE_TDM
#  define USE_TDM 0
#endif

#if USE_TDM
// Issue one TDM 2D tile load: rows x TK bf16 elements, row stride = K elems,
// into LDS at lds_off with 16B padding appended per 64B row (pad_interval=16DW,
// pad_amount=4DW -> LDS row stride 80B == LDS_STRIDE elems).
__device__ __forceinline__ void tdm_load_tile(const __bf16* g, unsigned lds_off,
                                              int rows, int K) {
  unsigned long long ga = (unsigned long long)(const void*)g;
  v4u g0;
  g0[0] = 1u;                                       // count=1 (user descriptor)
  g0[1] = lds_off;                                  // LDS byte address
  g0[2] = (unsigned)(ga & 0xFFFFFFFFu);             // global_addr[31:0]
  g0[3] = (unsigned)((ga >> 32) & 0x1FFFFFFu)       // global_addr[56:32]
          | (2u << 30);                             // type=2 ("image")
  v8i g1;
  unsigned td0 = (unsigned)K;                       // tensor_dim0 (>= tile_dim0)
  unsigned td1 = (unsigned)rows;                    // tensor_dim1
  g1[0] = (int)((1u << 16)                          // data_size = 2 bytes
              | (1u << 20)                          // pad_enable
              | (3u << 22)                          // pad_interval: 16 DWORDs
              | (3u << 25));                        // pad_amount:   4 DWORDs
  g1[1] = (int)((td0 & 0xFFFFu) << 16);             // [63:48] tensor_dim0 lo16
  g1[2] = (int)((td0 >> 16) | ((td1 & 0xFFFFu) << 16));
  g1[3] = (int)((td1 >> 16) | ((unsigned)TK << 16));// [127:112] tile_dim0
  g1[4] = rows & 0xFFFF;                            // [143:128] tile_dim1
  g1[5] = K;                                        // tensor_dim0_stride[31:0]
  g1[6] = 0;
  g1[7] = 0;
  v4i z4 = {};
#if __clang_major__ >= 23
  v8i z8 = {};
  __builtin_amdgcn_tensor_load_to_lds(g0, g1, z4, z4, z8, 0);
#else
  __builtin_amdgcn_tensor_load_to_lds(g0, g1, z4, z4, 0);
#endif
}
#endif

// Stage one K-tile of A and Bt into the given LDS buffers.
__device__ __forceinline__ void stage_tiles(
    const __bf16* __restrict__ A, const __bf16* __restrict__ Bt,
    int tile_m, int tile_n, int K, int kt,
    __bf16* AsBuf, __bf16* BsBuf, int tid)
{
#if USE_TDM
  if (tid < 32) {   // one wave issues both DMAs (wave-uniform descriptors)
    tdm_load_tile(A  + (size_t)tile_m * K + kt * TK,
                  (unsigned)(size_t)(void*)AsBuf, TM, K);
    tdm_load_tile(Bt + (size_t)tile_n * K + kt * TK,
                  (unsigned)(size_t)(void*)BsBuf, TN, K);
  }
#else
  const int lrow  = tid >> 1;
  const int lhalf = tid & 1;
  const int k0 = kt * TK;
  {
    const __bf16* src = A + (size_t)(tile_m + lrow) * K + k0 + lhalf * 16;
    v8bf a0 = *(const v8bf*)src;
    v8bf a1 = *(const v8bf*)(src + 8);
    __bf16* dst = AsBuf + lrow * LDS_STRIDE + lhalf * 16;
    *(v8bf*)dst       = a0;
    *(v8bf*)(dst + 8) = a1;
  }
  {
    const __bf16* src = Bt + (size_t)(tile_n + lrow) * K + k0 + lhalf * 16;
    v8bf b0 = *(const v8bf*)src;
    v8bf b1 = *(const v8bf*)(src + 8);
    __bf16* dst = BsBuf + lrow * LDS_STRIDE + lhalf * 16;
    *(v8bf*)dst       = b0;
    *(v8bf*)(dst + 8) = b1;
  }
  __builtin_prefetch(A  + (size_t)(tile_m + lrow) * K + k0 + TK, 0, 0);
  __builtin_prefetch(Bt + (size_t)(tile_n + lrow) * K + k0 + TK, 0, 0);
#endif
}

// Build a 16-elem bf16 fragment from two 16B LDS chunks (ds_load_b128 x2).
__device__ __forceinline__ v16bf ld_frag(const __bf16* p, int off2) {
  v8bf lo = *(const v8bf*)p;
  v8bf hi = *(const v8bf*)(p + off2);
  v16bf r;
#pragma unroll
  for (int e = 0; e < 8; ++e) { r[e] = lo[e]; r[e + 8] = hi[e]; }
  return r;
}

// ---------------------------------------------------------------------------
// Tiled WMMA GEMM: C[M,N] = A[M,K] (bf16, row-major) * Bt[N,K]^T (bf16)
// 128x128 tile per block, 8 waves, each wave owns 64x32 (4x2 subtiles).
// Double-buffered LDS; tiles staged by the Tensor Data Mover (TENSORcnt).
// ---------------------------------------------------------------------------
template<int EPI>
__global__ __launch_bounds__(256) void gemm_wmma_kernel(
    const __bf16* __restrict__ A, const __bf16* __restrict__ Bt,
    float* __restrict__ Cf, __bf16* __restrict__ Cb,
    const float* __restrict__ res, int M, int N, int K)
{
  __shared__ __bf16 As[2][TM * LDS_STRIDE];
  __shared__ __bf16 Bs[2][TN * LDS_STRIDE];

  const int tile_n = blockIdx.x * TN;
  const int tile_m = blockIdx.y * TM;
  const int tid  = threadIdx.x;
  const int wave = tid >> 5;
  const int lane = tid & 31;
  const int laneRow  = lane & 15;
  const int laneHalf = lane >> 4;
  const int wm = wave & 1;   // 2 waves across M (64 rows each)
  const int wn = wave >> 1;  // 4 waves across N (32 cols each)

  v8f acc[4][2] = {};

  const int KT = K / TK;

  // Pipeline prologue: stage tile 0 into buffer 0.
  stage_tiles(A, Bt, tile_m, tile_n, K, 0, &As[0][0], &Bs[0][0], tid);

  for (int kt = 0; kt < KT; ++kt) {
    // All waves done reading buf[(kt+1)&1] (read in the previous iteration) before
    // the TDM overwrites it with tile kt+1.
    __syncthreads();
    if (kt + 1 < KT)
      stage_tiles(A, Bt, tile_m, tile_n, K, kt + 1,
                  &As[(kt + 1) & 1][0], &Bs[(kt + 1) & 1][0], tid);
#if USE_TDM
    if (tid < 32) {
      // TDM ops complete in order: leave the 2 just-issued ops outstanding,
      // guarantee tile kt's pair has landed.
      if (kt + 1 < KT) __builtin_amdgcn_s_wait_tensorcnt((short)2);
      else             __builtin_amdgcn_s_wait_tensorcnt((short)0);
    }
#endif
    __syncthreads();

    const __bf16* Ab = &As[kt & 1][0];
    const __bf16* Bb = &Bs[kt & 1][0];

    v16bf af[4], bfr[2];
    // A fragment per ISA layout: lanes0-15 -> K {0..7,16..23}; lanes16-31 -> {8..15,24..31}
#pragma unroll
    for (int i = 0; i < 4; ++i) {
      int row = wm * 64 + i * 16 + laneRow;
      af[i] = ld_frag(Ab + row * LDS_STRIDE + laneHalf * 8, 16);
    }
    // B fragment (same as A): lanes0-15 -> K {0..7,16..23}; lanes16-31 -> {8..15,24..31}
#pragma unroll
    for (int j = 0; j < 2; ++j) {
      int col = wn * 32 + j * 16 + laneRow;
      bfr[j] = ld_frag(Bb + col * LDS_STRIDE + laneHalf * 8, 16);
    }
#pragma unroll
    for (int i = 0; i < 4; ++i)
#pragma unroll
      for (int j = 0; j < 2; ++j)
        acc[i][j] = __builtin_amdgcn_wmma_f32_16x16x32_bf16(
            false, af[i], false, bfr[j], (short)0, acc[i][j], false, false);
  }

  // Epilogue. C layout: VGPR r, lanes0-15 -> (M = 8*laneHalf + r, N = laneRow).
#pragma unroll
  for (int i = 0; i < 4; ++i) {
#pragma unroll
    for (int j = 0; j < 2; ++j) {
      int n = tile_n + wn * 32 + j * 16 + laneRow;
#pragma unroll
      for (int r = 0; r < 8; ++r) {
        int m = tile_m + wm * 64 + i * 16 + laneHalf * 8 + r;
        size_t idx = (size_t)m * N + n;
        float val = acc[i][j][r];
        if (EPI == EPI_ADD_F32) {
          Cf[idx] = val + res[idx];
        } else if (EPI == EPI_GELU_BF16) {
          float gel = 0.5f * val * (1.0f + erff(val * 0.70710678118654752f));
          Cb[idx] = (__bf16)gel;
        } else {
          Cf[idx] = val;
        }
      }
    }
  }
}

// ---------------------------------------------------------------------------
// Weight transpose + fp32->bf16 convert:  Wt[c][r] = (bf16)W[r][c]
// ---------------------------------------------------------------------------
__global__ __launch_bounds__(256) void transpose_cvt_kernel(
    const float* __restrict__ W, __bf16* __restrict__ Wt, int R, int C)
{
  __shared__ float tile[32][33];
  const int c0 = blockIdx.x * 32;
  const int r0 = blockIdx.y * 32;
  const int tx = threadIdx.x & 31;
  const int ty = threadIdx.x >> 5;   // 0..7
#pragma unroll
  for (int i = ty; i < 32; i += 8)
    tile[i][tx] = W[(size_t)(r0 + i) * C + c0 + tx];
  __syncthreads();
#pragma unroll
  for (int i = ty; i < 32; i += 8)
    Wt[(size_t)(c0 + i) * R + r0 + tx] = (__bf16)tile[tx][i];
}

// ---------------------------------------------------------------------------
// LayerNorm (D=1024) -> bf16
// ---------------------------------------------------------------------------
__global__ __launch_bounds__(256) void layernorm_bf16_kernel(
    const float* __restrict__ x, const float* __restrict__ g,
    __bf16* __restrict__ h, int D)
{
  __shared__ float rs[256];
  __shared__ float rss[256];
  const int tid = threadIdx.x;
  const size_t base = (size_t)blockIdx.x * D;

  float vals[4];
  float s = 0.f, ss = 0.f;
#pragma unroll
  for (int i = 0; i < 4; ++i) {
    float v = x[base + tid + i * 256];
    vals[i] = v; s += v; ss += v * v;
  }
  rs[tid] = s; rss[tid] = ss;
  __syncthreads();
  for (int off = 128; off > 0; off >>= 1) {
    if (tid < off) { rs[tid] += rs[tid + off]; rss[tid] += rss[tid + off]; }
    __syncthreads();
  }
  const float mean = rs[0] * (1.0f / 1024.0f);
  const float var  = rss[0] * (1.0f / 1024.0f) - mean * mean;
  const float rstd = rsqrtf(var + 1e-5f);
#pragma unroll
  for (int i = 0; i < 4; ++i) {
    int col = tid + i * 256;
    h[base + col] = (__bf16)((vals[i] - mean) * rstd * g[col]);
  }
}

// ---------------------------------------------------------------------------
// In-place radix-2 DIT FFT, N=1024, 256 threads, data in LDS.
// Input must be written bit-reversed; output is natural order.
// ---------------------------------------------------------------------------
__device__ __forceinline__ void fft1024(float* re, float* im, int tid) {
#pragma unroll
  for (int s = 1; s <= 10; ++s) {
    const int m = 1 << s;
    const int half = m >> 1;
    __syncthreads();
#pragma unroll
    for (int it = 0; it < 2; ++it) {
      int idx = tid + (it << 8);         // 0..511 butterflies
      int grp = idx / half;
      int pos = idx - grp * half;
      int i0 = grp * m + pos;
      int i1 = i0 + half;
      float ang = -6.283185307179586f * (float)pos / (float)m;
      float sn, cs;
      __sincosf(ang, &sn, &cs);
      float xr = re[i1], xi = im[i1];
      float tr = cs * xr - sn * xi;
      float ti = cs * xi + sn * xr;
      float ar = re[i0], ai = im[i0];
      re[i1] = ar - tr; im[i1] = ai - ti;
      re[i0] = ar + tr; im[i0] = ai + ti;
    }
  }
  __syncthreads();
}

__device__ __forceinline__ int brev10(int j) {
  return (int)(__brev((unsigned)j) >> 22);
}

// FFT(k)*FFT(v) elementwise (HRR bind) -> P planes
__global__ __launch_bounds__(256) void fft_bind_kernel(
    const float* __restrict__ k, const float* __restrict__ v,
    float* __restrict__ Pre, float* __restrict__ Pim)
{
  __shared__ float kr[1024], ki[1024], vr[1024], vi[1024];
  const int tid = threadIdx.x;
  const size_t base = (size_t)blockIdx.x * 1024;
#pragma unroll
  for (int i = 0; i < 4; ++i) {
    int j = tid + i * 256;
    int bj = brev10(j);
    kr[bj] = k[base + j]; ki[bj] = 0.f;
    vr[bj] = v[base + j]; vi[bj] = 0.f;
  }
  fft1024(kr, ki, tid);
  fft1024(vr, vi, tid);
#pragma unroll
  for (int i = 0; i < 4; ++i) {
    int j = tid + i * 256;
    Pre[base + j] = kr[j] * vr[j] - ki[j] * vi[j];
    Pim[base + j] = kr[j] * vi[j] + ki[j] * vr[j];
  }
}

// Forward FFT of real q -> Q planes
__global__ __launch_bounds__(256) void fft_fwd_kernel(
    const float* __restrict__ q, float* __restrict__ Qre, float* __restrict__ Qim)
{
  __shared__ float zr[1024], zi[1024];
  const int tid = threadIdx.x;
  const size_t base = (size_t)blockIdx.x * 1024;
#pragma unroll
  for (int i = 0; i < 4; ++i) {
    int j = tid + i * 256;
    int bj = brev10(j);
    zr[bj] = q[base + j]; zi[bj] = 0.f;
  }
  fft1024(zr, zi, tid);
#pragma unroll
  for (int i = 0; i < 4; ++i) {
    int j = tid + i * 256;
    Qre[base + j] = zr[j];
    Qim[base + j] = zi[j];
  }
}

// Causal cumsum over sequence, in place on one plane [B,S,D]
__global__ __launch_bounds__(256) void cumsum_seq_kernel(
    float* __restrict__ P, int S, int D)
{
  const int chunks = D / 256;
  const int b = blockIdx.x / chunks;
  const int dbase = (blockIdx.x % chunks) * 256;
  const int col = dbase + threadIdx.x;
  size_t base = (size_t)b * S * D + col;
  float acc = 0.f;
  for (int t = 0; t < S; ++t) {
    size_t idx = base + (size_t)t * D;
    acc += P[idx];
    P[idx] = acc;
  }
}

// z = mem * conj(fq); out = real(IFFT(z)) via conj trick; store bf16
__global__ __launch_bounds__(256) void unbind_ifft_kernel(
    const float* __restrict__ Pre, const float* __restrict__ Pim,
    const float* __restrict__ Qre, const float* __restrict__ Qim,
    __bf16* __restrict__ u)
{
  __shared__ float zr[1024], zi[1024];
  const int tid = threadIdx.x;
  const size_t base = (size_t)blockIdx.x * 1024;
#pragma unroll
  for (int i = 0; i < 4; ++i) {
    int j = tid + i * 256;
    float pr = Pre[base + j], pi = Pim[base + j];
    float qr = Qre[base + j], qi = Qim[base + j];
    float re = pr * qr + pi * qi;       // mem * conj(q)
    float im = pi * qr - pr * qi;
    int bj = brev10(j);
    zr[bj] = re;
    zi[bj] = -im;                       // conj for inverse transform
  }
  fft1024(zr, zi, tid);
#pragma unroll
  for (int i = 0; i < 4; ++i) {
    int j = tid + i * 256;
    u[base + j] = (__bf16)(zr[j] * (1.0f / 1024.0f));
  }
}

// ---------------------------------------------------------------------------
// Launcher
// ---------------------------------------------------------------------------
extern "C" void kernel_launch(void* const* d_in, const int* in_sizes, int n_in,
                              void* d_out, int out_size, void* d_ws, size_t ws_size,
                              hipStream_t stream)
{
  (void)in_sizes; (void)n_in; (void)out_size; (void)ws_size;
  const float* x  = (const float*)d_in[0];
  const float* Wq = (const float*)d_in[1];
  const float* Wk = (const float*)d_in[2];
  const float* Wv = (const float*)d_in[3];
  const float* Wo = (const float*)d_in[4];
  const float* g1 = (const float*)d_in[5];
  const float* g2 = (const float*)d_in[6];
  const float* W1 = (const float*)d_in[7];
  const float* W2 = (const float*)d_in[8];
  float* out = (float*)d_out;

  const int Bb = 4, S = 2048, D = 1024, F = 4096;
  const int ROWS = Bb * S;                  // 8192

  char* ws = (char*)d_ws;
  size_t off = 0;
  auto alloc = [&](size_t bytes) -> void* {
    off = (off + 255) & ~(size_t)255;
    void* p = ws + off;
    off += bytes;
    return p;
  };

  __bf16* hb  = (__bf16*)alloc((size_t)ROWS * D * 2);
  __bf16* Wtq = (__bf16*)alloc((size_t)D * D * 2);
  __bf16* Wtk = (__bf16*)alloc((size_t)D * D * 2);
  __bf16* Wtv = (__bf16*)alloc((size_t)D * D * 2);
  __bf16* Wto = (__bf16*)alloc((size_t)D * D * 2);
  __bf16* Wt1 = (__bf16*)alloc((size_t)F * D * 2);
  __bf16* Wt2 = (__bf16*)alloc((size_t)D * F * 2);
  float*  qf  = (float*)alloc((size_t)ROWS * D * 4);
  float*  kf  = (float*)alloc((size_t)ROWS * D * 4);
  float*  vf  = (float*)alloc((size_t)ROWS * D * 4);
  float*  Pre = (float*)alloc((size_t)ROWS * D * 4);
  float*  Pim = (float*)alloc((size_t)ROWS * D * 4);
  float*  Qre = (float*)alloc((size_t)ROWS * D * 4);
  float*  Qim = (float*)alloc((size_t)ROWS * D * 4);
  __bf16* u   = (__bf16*)alloc((size_t)ROWS * D * 2);
  float*  x2  = (float*)alloc((size_t)ROWS * D * 4);
  __bf16* h2  = (__bf16*)alloc((size_t)ROWS * D * 2);
  __bf16* act = (__bf16*)alloc((size_t)ROWS * F * 2);

  dim3 blk(256);

  // Weights -> bf16 transposed [N][K]
  transpose_cvt_kernel<<<dim3(D / 32, D / 32), blk, 0, stream>>>(Wq, Wtq, D, D);
  transpose_cvt_kernel<<<dim3(D / 32, D / 32), blk, 0, stream>>>(Wk, Wtk, D, D);
  transpose_cvt_kernel<<<dim3(D / 32, D / 32), blk, 0, stream>>>(Wv, Wtv, D, D);
  transpose_cvt_kernel<<<dim3(D / 32, D / 32), blk, 0, stream>>>(Wo, Wto, D, D);
  transpose_cvt_kernel<<<dim3(F / 32, D / 32), blk, 0, stream>>>(W1, Wt1, D, F);
  transpose_cvt_kernel<<<dim3(D / 32, F / 32), blk, 0, stream>>>(W2, Wt2, F, D);

  // Pre-norm
  layernorm_bf16_kernel<<<ROWS, blk, 0, stream>>>(x, g1, hb, D);

  // q, k, v projections (WMMA)
  dim3 gdd(D / TN, ROWS / TM);
  gemm_wmma_kernel<EPI_F32><<<gdd, blk, 0, stream>>>(hb, Wtq, qf, nullptr, nullptr, ROWS, D, D);
  gemm_wmma_kernel<EPI_F32><<<gdd, blk, 0, stream>>>(hb, Wtk, kf, nullptr, nullptr, ROWS, D, D);
  gemm_wmma_kernel<EPI_F32><<<gdd, blk, 0, stream>>>(hb, Wtv, vf, nullptr, nullptr, ROWS, D, D);

  // HRR: bind in frequency domain, causal memory, unbind
  fft_bind_kernel<<<ROWS, blk, 0, stream>>>(kf, vf, Pre, Pim);
  fft_fwd_kernel<<<ROWS, blk, 0, stream>>>(qf, Qre, Qim);
  cumsum_seq_kernel<<<Bb * (D / 256), blk, 0, stream>>>(Pre, S, D);
  cumsum_seq_kernel<<<Bb * (D / 256), blk, 0, stream>>>(Pim, S, D);
  unbind_ifft_kernel<<<ROWS, blk, 0, stream>>>(Pre, Pim, Qre, Qim, u);

  // Output projection + residual
  gemm_wmma_kernel<EPI_ADD_F32><<<gdd, blk, 0, stream>>>(u, Wto, x2, nullptr, x, ROWS, D, D);

  // MLP sublayer
  layernorm_bf16_kernel<<<ROWS, blk, 0, stream>>>(x2, g2, h2, D);
  gemm_wmma_kernel<EPI_GELU_BF16><<<dim3(F / TN, ROWS / TM), blk, 0, stream>>>(
      h2, Wt1, nullptr, act, nullptr, ROWS, F, D);
  gemm_wmma_kernel<EPI_ADD_F32><<<dim3(D / TN, ROWS / TM), blk, 0, stream>>>(
      act, Wt2, out, nullptr, x2, ROWS, D, F);
}
